// Block_57707180589710
// MI455X (gfx1250) — matrix operand order repaired and checked by reading, not compiled, not run on hardware
//
#include <hip/hip_runtime.h>
#include <hip/hip_bf16.h>

// ---------------------------------------------------------------------------
// MI455X (gfx1250) equivariant message-passing layer.
// Heavy per-edge GEMMs (16-edge row tiles) run on v_wmma_f32_16x16x32_f16.
// ---------------------------------------------------------------------------

#define DIMA   128
#define DIMV   32
#define NCHAN  32
#define LEAKC  0.1f
#define RZERO  5.0f
#define WAVES  4          // waves (16-edge tiles) per block

typedef __attribute__((ext_vector_type(16))) _Float16 v16h;
typedef __attribute__((ext_vector_type(8)))  float    v8f;

union FragA { v16h h; unsigned int u[8]; };
union FragB { v16h h; uint4 q[2]; };

// Per-wave LDS staging (one 16-edge tile per wave).
struct WaveLds {
  float    rad[16][8];     // radial basis per edge
  float    rs[16][4];      // squashed direction (padded)
  float    phi[16][NCHAN]; // rad_enc (f32)
  float    lv[16][NCHAN * 3];
  int      srcI[16];
  int      dstI[16];
  int      valid[16];
  _Float16 y000[16 * NCHAN];
  _Float16 y110[16 * NCHAN];
  _Float16 stage[16 * DIMA]; // MLP activation re-layout / psi_v A staging
};

// Wave-scope LDS ordering: LDS ops are in-order per wave, but we must stop the
// compiler from reordering and force completion of outstanding DS ops before
// cross-lane reads. CDNA5 split counter: DScnt.
__device__ __forceinline__ void wave_lds_sync() {
  __builtin_amdgcn_wave_barrier();
  asm volatile("s_wait_dscnt 0" ::: "memory");
  __builtin_amdgcn_wave_barrier();
}

// f16 A-matrix 16x32 fragment from a row-major [16][stride] LDS buffer.
// ISA §7.12.2: lane m=l%16, block=l/16; VGPR v<4: K=2v+p+8*blk, v>=4: K=16+2(v-4)+p+8*blk.
// K pairs are contiguous -> 8x 32-bit LDS reads.
__device__ __forceinline__ v16h load_a_frag(const _Float16* buf, int stride, int lane) {
  const int m = lane & 15, blk = lane >> 4;
  const _Float16* row = buf + m * stride;
  FragA f;
#pragma unroll
  for (int v = 0; v < 8; ++v) {
    const int kb = (v < 4) ? (2 * v + 8 * blk) : (16 + 2 * (v - 4) + 8 * blk);
    f.u[v] = *reinterpret_cast<const unsigned int*>(row + kb);
  }
  return f.h;
}

// Pre-packed B fragment: 512 halves per fragment, lane-major, 32B per lane.
__device__ __forceinline__ v16h load_b_frag(const _Float16* pack, int frag, int lane) {
  const uint4* g = reinterpret_cast<const uint4*>(pack + (size_t)frag * 512 + lane * 16);
  FragB f;
  f.q[0] = g[0];
  f.q[1] = g[1];
  return f.h;
}

__device__ __forceinline__ v8f wmma_f16(v16h a, v16h b, v8f c) {
  return __builtin_amdgcn_wmma_f32_16x16x32_f16(false, a, false, b, (short)0, c,
                                                false, false);
}

__device__ __forceinline__ v8f leaky8(v8f x) {
#pragma unroll
  for (int r = 0; r < 8; ++r) x[r] = (x[r] >= 0.f) ? x[r] : LEAKC * x[r];
  return x;
}

// ---------------------------------------------------------------------------
// Weight packer: W is row-major (OUT, IN). B fragment (K=in, N=out):
// lane l: n=l%16, blk=l/16; slot s: v=s>>1, p=s&1, k_local = (v<4) ? 2v+p+8*blk : 16+2(v-4)+p+8*blk.
// ---------------------------------------------------------------------------
__global__ void pack_weight_kernel(const float* __restrict__ W,
                                   _Float16* __restrict__ dst, int OUT, int IN) {
  const int ktiles = IN / 32;
  const int frag = blockIdx.x;
  const int nt = frag / ktiles;
  const int kt = frag % ktiles;
  for (int s = threadIdx.x; s < 512; s += blockDim.x) {
    const int lane = s >> 4, slot = s & 15;
    const int n = nt * 16 + (lane & 15);
    const int blk = lane >> 4;
    const int v = slot >> 1, p = slot & 1;
    const int k = kt * 32 + ((v < 4) ? (2 * v + p + 8 * blk) : (16 + 2 * (v - 4) + p + 8 * blk));
    dst[(size_t)frag * 512 + s] = (_Float16)W[(size_t)n * IN + k];
  }
}

__global__ void zero_kernel(float* __restrict__ out, long n) {
  long g = (long)blockIdx.x * blockDim.x + threadIdx.x;
  if (g < n) out[g] = 0.f;
}

// la[n][c] = sum_d x_a[n][d] * W_L0[c][d]
__global__ void proj_la_kernel(const float* __restrict__ x_a,
                               const float* __restrict__ W_L0,
                               float* __restrict__ la, int N) {
  long g = (long)blockIdx.x * blockDim.x + threadIdx.x;
  if (g >= (long)N * NCHAN) return;
  const int n = (int)(g >> 5), c = (int)(g & 31);
  const float* xr = x_a + (long)n * DIMA;
  const float* wr = W_L0 + c * DIMA;
  float s = 0.f;
#pragma unroll 8
  for (int d = 0; d < DIMA; ++d) s += xr[d] * wr[d];
  la[g] = s;
}

// lv[n][c][i] = sum_d W_L1[c][d] * x_v[n][d][i]
__global__ void proj_lv_kernel(const float* __restrict__ x_v,
                               const float* __restrict__ W_L1,
                               float* __restrict__ lv, int N) {
  long g = (long)blockIdx.x * blockDim.x + threadIdx.x;
  if (g >= (long)N * NCHAN * 3) return;
  const int n = (int)(g / 96);
  const int rem = (int)(g % 96);
  const int c = rem / 3, i = rem % 3;
  const float* xr = x_v + (long)n * 96;
  const float* wr = W_L1 + c * NCHAN;
  float s = 0.f;
#pragma unroll 8
  for (int d = 0; d < NCHAN; ++d) s += wr[d] * xr[d * 3 + i];
  lv[g] = s;
}

// ---------------------------------------------------------------------------
// Main per-edge kernel: one 16-edge tile per wave, 4 waves per block.
// ---------------------------------------------------------------------------
__global__ __launch_bounds__(WAVES * 32)
void edge_kernel(const float* __restrict__ r_ij, const int* __restrict__ srcg,
                 const int* __restrict__ dstg,
                 const float* __restrict__ W_enc, const float* __restrict__ b_enc,
                 const float* __restrict__ bm1, const float* __restrict__ bm2,
                 const float* __restrict__ la, const float* __restrict__ lvn,
                 const _Float16* __restrict__ pWy000, const _Float16* __restrict__ pWy110,
                 const _Float16* __restrict__ pWm1, const _Float16* __restrict__ pWm2,
                 const _Float16* __restrict__ pWm3,
                 const _Float16* __restrict__ pWy011, const _Float16* __restrict__ pWy101,
                 const _Float16* __restrict__ pWy111,
                 float* __restrict__ Ba, float* __restrict__ Bv, int E) {
  __shared__ WaveLds smem[WAVES];
  const int lane = threadIdx.x & 31;
  const int wave = threadIdx.x >> 5;
  WaveLds& S = smem[wave];
  const long base = ((long)blockIdx.x * WAVES + wave) * 16;
  const int nl = lane & 15, blk = lane >> 4;

  // ---- phase 0: per-edge scalars (lanes 0..15 own one edge each) -----------
  if (lane < 16) {
    const long e = base + lane;
    const int ok = (e < (long)E);
    const long ec = ok ? e : (long)(E - 1);
    const float x = r_ij[ec * 3 + 0], y = r_ij[ec * 3 + 1], z = r_ij[ec * 3 + 2];
    const float rn = sqrtf(x * x + y * y + z * z);
    const float cstep = RZERO / 7.0f;   // linspace(0, R0, 8)
    const float iw = 8.0f / RZERO;      // 1/width
#pragma unroll
    for (int j = 0; j < 8; ++j) {
      const float d = (rn - cstep * (float)j) * iw;
      S.rad[lane][j] = __expf(-d * d);
    }
    const float sn = rn * 1.4f;  // |r_ij * 7/R0|
    const float t = tanhf(sn) / fmaxf(sn, 1e-6f);
    S.rs[lane][0] = x * 1.4f * t;
    S.rs[lane][1] = y * 1.4f * t;
    S.rs[lane][2] = z * 1.4f * t;
    S.srcI[lane] = srcg[ec];
    S.dstI[lane] = dstg[ec];
    S.valid[lane] = ok;
  }
  wave_lds_sync();

  // ---- phase 1: rad_enc + gathers + rank-0 products (lane = channel c) -----
  {
    float wrow[8];
#pragma unroll
    for (int j = 0; j < 8; ++j) wrow[j] = W_enc[lane * 8 + j];
    const float bc = b_enc[lane];
#pragma unroll 1
    for (int t = 0; t < 16; ++t) {
      float p = bc;
#pragma unroll
      for (int j = 0; j < 8; ++j) p += wrow[j] * S.rad[t][j];
      const int dn = S.dstI[t];
      const float laa = la[(long)dn * NCHAN + lane];
      const float l0 = lvn[(long)dn * 96 + lane * 3 + 0];
      const float l1 = lvn[(long)dn * 96 + lane * 3 + 1];
      const float l2 = lvn[(long)dn * 96 + lane * 3 + 2];
      const float r0 = S.rs[t][0], r1 = S.rs[t][1], r2 = S.rs[t][2];
      S.phi[t][lane] = p;
      S.lv[t][lane * 3 + 0] = l0;
      S.lv[t][lane * 3 + 1] = l1;
      S.lv[t][lane * 3 + 2] = l2;
      S.y000[t * NCHAN + lane] = (_Float16)(laa * p);
      S.y110[t * NCHAN + lane] = (_Float16)(p * (l0 * r0 + l1 * r1 + l2 * r2));
    }
  }
  wave_lds_sync();

  // ---- GEMM 1: psi_a = y000*Wy000^T + y110*Wy110^T  (16x32 @ 32x128) -------
  v8f psi[8];
  {
    const v16h a0 = load_a_frag(S.y000, NCHAN, lane);
    const v16h a1 = load_a_frag(S.y110, NCHAN, lane);
#pragma unroll
    for (int nt = 0; nt < 8; ++nt) {
      v8f acc = {0.f, 0.f, 0.f, 0.f, 0.f, 0.f, 0.f, 0.f};
      acc = wmma_f16(a0, load_b_frag(pWy000, nt, lane), acc);
      acc = wmma_f16(a1, load_b_frag(pWy110, nt, lane), acc);
      psi[nt] = acc;
    }
  }

  // Re-layout accumulator rows (C layout: m = r + 8*blk, n = nt*16 + nl).
  auto store_rows = [&](const v8f* v) {
#pragma unroll
    for (int nt = 0; nt < 8; ++nt)
#pragma unroll
      for (int r = 0; r < 8; ++r)
        S.stage[(r + 8 * blk) * DIMA + nt * 16 + nl] = (_Float16)v[nt][r];
  };

  store_rows(psi);
  wave_lds_sync();

  // ---- MLP layer 1: h = leaky(psi*Wm1^T + bm1) ------------------------------
  v8f h[8];
  {
    v16h af[4];
#pragma unroll
    for (int kt = 0; kt < 4; ++kt) af[kt] = load_a_frag(S.stage + kt * 32, DIMA, lane);
#pragma unroll
    for (int nt = 0; nt < 8; ++nt) {
      const float b = bm1[nt * 16 + nl];
      v8f acc = {b, b, b, b, b, b, b, b};
#pragma unroll
      for (int kt = 0; kt < 4; ++kt)
        acc = wmma_f16(af[kt], load_b_frag(pWm1, nt * 4 + kt, lane), acc);
      h[nt] = leaky8(acc);
    }
  }
  wave_lds_sync();
  store_rows(h);
  wave_lds_sync();

  // ---- MLP layer 2 ---------------------------------------------------------
  {
    v16h af[4];
#pragma unroll
    for (int kt = 0; kt < 4; ++kt) af[kt] = load_a_frag(S.stage + kt * 32, DIMA, lane);
#pragma unroll
    for (int nt = 0; nt < 8; ++nt) {
      const float b = bm2[nt * 16 + nl];
      v8f acc = {b, b, b, b, b, b, b, b};
#pragma unroll
      for (int kt = 0; kt < 4; ++kt)
        acc = wmma_f16(af[kt], load_b_frag(pWm2, nt * 4 + kt, lane), acc);
      h[nt] = leaky8(acc);
    }
  }
  wave_lds_sync();
  store_rows(h);
  wave_lds_sync();

  // ---- MLP layer 3 (residual: accumulate directly into psi) ----------------
  {
    v16h af[4];
#pragma unroll
    for (int kt = 0; kt < 4; ++kt) af[kt] = load_a_frag(S.stage + kt * 32, DIMA, lane);
#pragma unroll
    for (int nt = 0; nt < 8; ++nt)
#pragma unroll
      for (int kt = 0; kt < 4; ++kt)
        psi[nt] = wmma_f16(af[kt], load_b_frag(pWm3, nt * 4 + kt, lane), psi[nt]);
  }

  // ---- scatter psi_a -------------------------------------------------------
#pragma unroll
  for (int nt = 0; nt < 8; ++nt)
#pragma unroll
    for (int r = 0; r < 8; ++r) {
      const int m = r + 8 * blk;
      if (S.valid[m])
        atomicAdd(&Ba[(long)S.srcI[m] * DIMA + nt * 16 + nl], 0.1f * psi[nt][r]);
    }

  // ---- psi_v: per spatial dim i, (16x96) @ (96x32) split into 3 K-steps ----
#pragma unroll 1
  for (int i = 0; i < 3; ++i) {
    wave_lds_sync();  // protect stage reuse against previous phase's reads
    const int i1 = (i == 2) ? 0 : i + 1;
    const int i2 = (i1 == 2) ? 0 : i1 + 1;
#pragma unroll 1
    for (int t = 0; t < 16; ++t) {
      const float p = S.phi[t][lane];
      const float y000f = (float)S.y000[t * NCHAN + lane];
      const float lvi = S.lv[t][lane * 3 + i];
      const float lv1 = S.lv[t][lane * 3 + i1];
      const float lv2 = S.lv[t][lane * 3 + i2];
      const float rsi = S.rs[t][i];
      const float rs1 = S.rs[t][i1];
      const float rs2 = S.rs[t][i2];
      const float cr = lv1 * rs2 - lv2 * rs1;  // cross(lv, rs)_i
      S.stage[0 + t * NCHAN + lane] = (_Float16)(y000f * rsi);  // y011_i
      S.stage[512 + t * NCHAN + lane] = (_Float16)(p * lvi);    // y101_i
      S.stage[1024 + t * NCHAN + lane] = (_Float16)(p * cr);    // y111_i
    }
    wave_lds_sync();
    const v16h a011 = load_a_frag(S.stage + 0, NCHAN, lane);
    const v16h a101 = load_a_frag(S.stage + 512, NCHAN, lane);
    const v16h a111 = load_a_frag(S.stage + 1024, NCHAN, lane);
#pragma unroll
    for (int nt = 0; nt < 2; ++nt) {
      v8f acc = {0.f, 0.f, 0.f, 0.f, 0.f, 0.f, 0.f, 0.f};
      acc = wmma_f16(a011, load_b_frag(pWy011, nt, lane), acc);
      acc = wmma_f16(a101, load_b_frag(pWy101, nt, lane), acc);
      acc = wmma_f16(a111, load_b_frag(pWy111, nt, lane), acc);
#pragma unroll
      for (int r = 0; r < 8; ++r) {
        const int m = r + 8 * blk;
        if (S.valid[m])
          atomicAdd(&Bv[(long)S.srcI[m] * 96 + (nt * 16 + nl) * 3 + i],
                    0.1f * acc[r]);
      }
    }
  }
}

// ---------------------------------------------------------------------------
extern "C" void kernel_launch(void* const* d_in, const int* in_sizes, int n_in,
                              void* d_out, int out_size, void* d_ws, size_t ws_size,
                              hipStream_t stream) {
  const float* x_a  = (const float*)d_in[0];
  const float* x_v  = (const float*)d_in[1];
  const float* r_ij = (const float*)d_in[2];
  const int*   src  = (const int*)d_in[3];
  const int*   dst  = (const int*)d_in[4];
  const float* W_L0 = (const float*)d_in[5];
  const float* W_L1 = (const float*)d_in[6];
  const float* W_enc = (const float*)d_in[7];
  const float* b_enc = (const float*)d_in[8];
  const float* Wy000 = (const float*)d_in[9];
  const float* Wy110 = (const float*)d_in[10];
  const float* Wy011 = (const float*)d_in[11];
  const float* Wy101 = (const float*)d_in[12];
  const float* Wy111 = (const float*)d_in[13];
  const float* Wm1 = (const float*)d_in[14];
  const float* bm1 = (const float*)d_in[15];
  const float* Wm2 = (const float*)d_in[16];
  const float* bm2 = (const float*)d_in[17];
  const float* Wm3 = (const float*)d_in[18];

  const int N = in_sizes[0] / DIMA;     // 50000
  const int E = in_sizes[2] / 3;        // 500000

  float* out = (float*)d_out;
  float* Ba = out;
  float* Bv = out + (long)N * DIMA;

  // workspace layout (all 16B aligned)
  float* la = (float*)d_ws;
  float* lv = la + (size_t)N * NCHAN;
  _Float16* packs = (_Float16*)(lv + (size_t)N * NCHAN * 3);
  _Float16* pWy000 = packs;            // 8 frags
  _Float16* pWy110 = pWy000 + 8 * 512; // 8 frags
  _Float16* pWm1   = pWy110 + 8 * 512; // 32 frags
  _Float16* pWm2   = pWm1 + 32 * 512;
  _Float16* pWm3   = pWm2 + 32 * 512;
  _Float16* pWy011 = pWm3 + 32 * 512;  // 2 frags each
  _Float16* pWy101 = pWy011 + 2 * 512;
  _Float16* pWy111 = pWy101 + 2 * 512;

  const long outN = (long)out_size;
  zero_kernel<<<(int)((outN + 255) / 256), 256, 0, stream>>>(out, outN);

  {
    const long t = (long)N * NCHAN;
    proj_la_kernel<<<(int)((t + 255) / 256), 256, 0, stream>>>(x_a, W_L0, la, N);
  }
  {
    const long t = (long)N * NCHAN * 3;
    proj_lv_kernel<<<(int)((t + 255) / 256), 256, 0, stream>>>(x_v, W_L1, lv, N);
  }

  pack_weight_kernel<<<8, 128, 0, stream>>>(Wy000, pWy000, DIMA, NCHAN);
  pack_weight_kernel<<<8, 128, 0, stream>>>(Wy110, pWy110, DIMA, NCHAN);
  pack_weight_kernel<<<32, 128, 0, stream>>>(Wm1, pWm1, DIMA, DIMA);
  pack_weight_kernel<<<32, 128, 0, stream>>>(Wm2, pWm2, DIMA, DIMA);
  pack_weight_kernel<<<32, 128, 0, stream>>>(Wm3, pWm3, DIMA, DIMA);
  pack_weight_kernel<<<2, 128, 0, stream>>>(Wy011, pWy011, DIMV, NCHAN);
  pack_weight_kernel<<<2, 128, 0, stream>>>(Wy101, pWy101, DIMV, NCHAN);
  pack_weight_kernel<<<2, 128, 0, stream>>>(Wy111, pWy111, DIMV, NCHAN);

  const int tiles = (E + 15) / 16;
  const int blocks = (tiles + WAVES - 1) / WAVES;
  edge_kernel<<<blocks, WAVES * 32, 0, stream>>>(
      r_ij, src, dst, W_enc, b_enc, bm1, bm2, la, lv,
      pWy000, pWy110, pWm1, pWm2, pWm3, pWy011, pWy101, pWy111, Ba, Bv, E);

  (void)n_in; (void)ws_size;
}
